// RGTDetector_65283502899736
// MI455X (gfx1250) — matrix operand order repaired and checked by reading, not compiled, not run on hardware
//
#include <hip/hip_runtime.h>
#include <math.h>
#include <stdint.h>

// ---------------------------------------------------------------------------
// Types for CDNA5 WMMA
// ---------------------------------------------------------------------------
typedef __attribute__((ext_vector_type(16))) __bf16 v16bf;
typedef __attribute__((ext_vector_type(8)))  float  v8f;
typedef __attribute__((ext_vector_type(4)))  int    i4vec;
// pointer-to-int4 in global (as1) / LDS (as3) address spaces, for the
// CDNA5 async global->LDS copy builtins (signature probed via compile round).
typedef __attribute__((address_space(1))) i4vec* g_i4p;
typedef __attribute__((address_space(3))) i4vec* l_i4p;

#define NN 50000
#define EE 160000
#define DD 128

// CDNA5 async global->LDS copy (ASYNCcnt), if the toolchain exposes it.
#if defined(__gfx1250__) && \
    __has_builtin(__builtin_amdgcn_global_load_async_to_lds_b128) && \
    __has_builtin(__builtin_amdgcn_s_wait_asynccnt)
#define HAVE_ASYNC_LDS 1
#else
#define HAVE_ASYNC_LDS 0
#endif

__device__ __forceinline__ float bf2f(unsigned short h) {
  unsigned u = ((unsigned)h) << 16;
  return __uint_as_float(u);
}
__device__ __forceinline__ unsigned short f2bf(float f) {
  unsigned u = __float_as_uint(f);
  unsigned r = u + 0x7fffu + ((u >> 16) & 1u);   // round-to-nearest-even
  return (unsigned short)(r >> 16);
}
// monotone float<->uint encoding for atomicMax on floats
__device__ __forceinline__ unsigned fenc(float f) {
  unsigned u = __float_as_uint(f);
  return (u & 0x80000000u) ? ~u : (u | 0x80000000u);
}
__device__ __forceinline__ float fdec(unsigned e) {
  unsigned u = (e & 0x80000000u) ? (e & 0x7fffffffu) : ~e;
  return __uint_as_float(u);
}

__device__ __forceinline__ uint4 load8_as_bf16(const unsigned short* p) {
  return *(const uint4*)p;
}
__device__ __forceinline__ uint4 load8_as_bf16(const float* p) {
  unsigned short t[8];
#pragma unroll
  for (int i = 0; i < 8; ++i) t[i] = f2bf(p[i]);
  return *(uint4*)t;
}

// ---------------------------------------------------------------------------
// Tiled bf16 WMMA GEMM:  C[M x Nout] = act(A[M x K] @ W[K x Nout] + bias)
// Block tile 128x32, K step 64. 8 wave32 waves: wave (wm,wn) owns a 32x16
// output tile (two v8f accumulators) -> 4 v_wmma per K-iteration per wave.
// B is staged TRANSPOSED in LDS so both A and B fragments are 2x b128 loads.
// A: f32 or bf16 row-major (lda).  W: bf16 row-major (ldw = Nout).
// ACT: 0=none 1=lrelu 2=sigmoid 3=tanh.  OUTBF: write bf16 else f32.
// Requires: K % 64 == 0, Nout % 32 == 0 (true for all uses here).
// ---------------------------------------------------------------------------
#define BM 128
#define BN 32
#define BK 64

template <typename AT, int ACT, bool OUTBF>
__global__ __launch_bounds__(256) void gemm_wmma(
    const AT* __restrict__ A, int lda,
    const unsigned short* __restrict__ W, int ldw,
    const float* __restrict__ bias,
    void* __restrict__ Cout, int ldc, int colOff,
    int M, int K, int Nout) {
  __shared__ unsigned short As[BM * BK];   // [row][k]      128x64
  __shared__ unsigned short Bt[BN * BK];   // [col][k] (transposed) 32x64

  const int tid  = threadIdx.x;
  const int bm0  = blockIdx.x * BM;
  const int bn0  = blockIdx.y * BN;
  const int wave = tid >> 5;
  const int lane = tid & 31;
  const int wm   = wave >> 1;   // 0..3 -> rows wm*32 .. wm*32+31
  const int wn   = wave & 1;    // 0..1 -> cols wn*16 .. wn*16+15

  v8f acc0 = {};
  v8f acc1 = {};

  const int brow  = tid >> 2;          // 0..63 (B k-row)
  const int bcol8 = (tid & 3) * 8;     // 0,8,16,24 (B col chunk)

  for (int k0 = 0; k0 < K; k0 += BK) {
    // ---- stage A tile: 128x64 bf16, 4 chunks of 8 shorts per thread ----
#pragma unroll
    for (int j = 0; j < 4; ++j) {
      int chunk = tid + 256 * j;        // 0..1023
      int row   = chunk >> 3;           // 0..127
      int col8  = (chunk & 7) * 8;      // 0..56
      int gr    = bm0 + row;
      if (gr >= M) gr = M - 1;          // clamp: pad rows never stored
      const AT* src = A + (size_t)gr * lda + k0 + col8;
#if HAVE_ASYNC_LDS
      if (sizeof(AT) == 2) {
        __builtin_amdgcn_global_load_async_to_lds_b128(
            (g_i4p)(uintptr_t)src,
            (l_i4p)(uintptr_t)&As[row * BK + col8],
            0, 0);
      } else {
        *(uint4*)&As[row * BK + col8] = load8_as_bf16(src);
      }
#else
      *(uint4*)&As[row * BK + col8] = load8_as_bf16(src);
#endif
    }
    // ---- stage B tile transposed: Bt[col][k], one 8-short chunk/thread ----
    {
      uint4 v = *(const uint4*)(W + (size_t)(k0 + brow) * ldw + bn0 + bcol8);
      const unsigned short* s = (const unsigned short*)&v;
#pragma unroll
      for (int i = 0; i < 8; ++i) Bt[(bcol8 + i) * BK + brow] = s[i];
    }
#if HAVE_ASYNC_LDS
    if (sizeof(AT) == 2) __builtin_amdgcn_s_wait_asynccnt(0);
#endif
    __syncthreads();

    // ---- two K-halves of 32, two M-subtiles each ----
#pragma unroll
    for (int kh = 0; kh < 2; ++kh) {
      const int khalf = kh * 32;
      // B fragment: col = wn*16 + (lane&15); K = khalf + {0..7,16..23 | 8..15,24..31}
      union { v16bf v; uint4 q[2]; } bf;
      {
        int c  = wn * 16 + (lane & 15);
        int kb = (lane < 16) ? 0 : 8;
        const unsigned short* p = &Bt[c * BK + khalf + kb];
        bf.q[0] = *(const uint4*)p;       // K pairs (2k,2k+1) contiguous
        bf.q[1] = *(const uint4*)(p + 16);
      }
      // A fragments: lane<16 -> K{0..7,16..23}, lane>=16 -> K{8..15,24..31}
      union { v16bf v; uint4 q[2]; } af0, af1;
      {
        int ka = (lane < 16) ? 0 : 8;
        int r0 = wm * 32 + (lane & 15);
        const unsigned short* p0 = &As[r0 * BK + khalf + ka];
        af0.q[0] = *(const uint4*)p0;
        af0.q[1] = *(const uint4*)(p0 + 16);
        const unsigned short* p1 = &As[(r0 + 16) * BK + khalf + ka];
        af1.q[0] = *(const uint4*)p1;
        af1.q[1] = *(const uint4*)(p1 + 16);
      }
      acc0 = __builtin_amdgcn_wmma_f32_16x16x32_bf16(
          false, af0.v, false, bf.v, (short)0, acc0, false, false);
      acc1 = __builtin_amdgcn_wmma_f32_16x16x32_bf16(
          false, af1.v, false, bf.v, (short)0, acc1, false, false);
    }
    __syncthreads();
  }

  // ---- epilogue: bias + activation + store (C/D layout: N=lane&15,
  //      M = j + 8*(lane>=16)) ----
  int gc = bn0 + wn * 16 + (lane & 15);
  float bv = bias ? bias[gc] : 0.0f;
#pragma unroll
  for (int t = 0; t < 2; ++t) {
    v8f acc = t ? acc1 : acc0;
#pragma unroll
    for (int j = 0; j < 8; ++j) {
      int m  = j + ((lane >> 4) << 3);
      int gr = bm0 + wm * 32 + t * 16 + m;
      if (gr < M) {
        float v = acc[j] + bv;
        if (ACT == 1) v = (v > 0.f) ? v : 0.01f * v;
        if (ACT == 2) v = 1.0f / (1.0f + __expf(-v));
        if (ACT == 3) v = tanhf(v);
        if (OUTBF)
          ((unsigned short*)Cout)[(size_t)gr * ldc + colOff + gc] = f2bf(v);
        else
          ((float*)Cout)[(size_t)gr * ldc + colOff + gc] = v;
      }
    }
  }
}

// ---------------------------------------------------------------------------
// Fused semantic attention GEMM:
//   H = tanh(Z @ w1[s] + b1[s]);  wsc[s,m] += sum over rows of parity m of
//   (H . w2[s]).   Z is (2N)x128 bf16 (row = n*2 + m).  Same tiling.
// ---------------------------------------------------------------------------
__global__ __launch_bounds__(256) void gemm_sem(
    const unsigned short* __restrict__ Z,
    const unsigned short* __restrict__ W1,  // [8][128][128] bf16
    const float* __restrict__ B1,           // [8][128]
    const float* __restrict__ W2,           // [8][128]
    float* __restrict__ wsc,                // [8][2]
    int M) {
  __shared__ unsigned short As[BM * BK];
  __shared__ unsigned short Bt[BN * BK];

  const int tid  = threadIdx.x;
  const int bm0  = blockIdx.x * BM;
  const int bn0  = blockIdx.y * BN;
  const int s    = blockIdx.z;
  const int wave = tid >> 5;
  const int lane = tid & 31;
  const int wm   = wave >> 1;
  const int wn   = wave & 1;

  const unsigned short* W  = W1 + (size_t)s * 128 * 128;
  const float*          b1 = B1 + s * 128;
  const float*          w2 = W2 + s * 128;

  v8f acc0 = {};
  v8f acc1 = {};

  const int brow  = tid >> 2;
  const int bcol8 = (tid & 3) * 8;

  for (int k0 = 0; k0 < 128; k0 += BK) {
#pragma unroll
    for (int j = 0; j < 4; ++j) {
      int chunk = tid + 256 * j;
      int row   = chunk >> 3;
      int col8  = (chunk & 7) * 8;
      int gr    = bm0 + row;
      if (gr >= M) gr = M - 1;
      const unsigned short* src = Z + (size_t)gr * 128 + k0 + col8;
#if HAVE_ASYNC_LDS
      __builtin_amdgcn_global_load_async_to_lds_b128(
          (g_i4p)(uintptr_t)src,
          (l_i4p)(uintptr_t)&As[row * BK + col8],
          0, 0);
#else
      *(uint4*)&As[row * BK + col8] = *(const uint4*)src;
#endif
    }
    {
      uint4 v = *(const uint4*)(W + (size_t)(k0 + brow) * 128 + bn0 + bcol8);
      const unsigned short* sp = (const unsigned short*)&v;
#pragma unroll
      for (int i = 0; i < 8; ++i) Bt[(bcol8 + i) * BK + brow] = sp[i];
    }
#if HAVE_ASYNC_LDS
    __builtin_amdgcn_s_wait_asynccnt(0);
#endif
    __syncthreads();

#pragma unroll
    for (int kh = 0; kh < 2; ++kh) {
      const int khalf = kh * 32;
      union { v16bf v; uint4 q[2]; } bf;
      {
        int c  = wn * 16 + (lane & 15);
        int kb = (lane < 16) ? 0 : 8;
        const unsigned short* p = &Bt[c * BK + khalf + kb];
        bf.q[0] = *(const uint4*)p;
        bf.q[1] = *(const uint4*)(p + 16);
      }
      union { v16bf v; uint4 q[2]; } af0, af1;
      {
        int ka = (lane < 16) ? 0 : 8;
        int r0 = wm * 32 + (lane & 15);
        const unsigned short* p0 = &As[r0 * BK + khalf + ka];
        af0.q[0] = *(const uint4*)p0;
        af0.q[1] = *(const uint4*)(p0 + 16);
        const unsigned short* p1 = &As[(r0 + 16) * BK + khalf + ka];
        af1.q[0] = *(const uint4*)p1;
        af1.q[1] = *(const uint4*)(p1 + 16);
      }
      acc0 = __builtin_amdgcn_wmma_f32_16x16x32_bf16(
          false, af0.v, false, bf.v, (short)0, acc0, false, false);
      acc1 = __builtin_amdgcn_wmma_f32_16x16x32_bf16(
          false, af1.v, false, bf.v, (short)0, acc1, false, false);
    }
    __syncthreads();
  }

  // fused epilogue: tanh -> * w2[col] -> per-parity partial sums
  int gc   = bn0 + wn * 16 + (lane & 15);
  float bv = b1[gc];
  float wv = w2[gc];
  float se = 0.f, so = 0.f;
#pragma unroll
  for (int t = 0; t < 2; ++t) {
    v8f acc = t ? acc1 : acc0;
#pragma unroll
    for (int j = 0; j < 8; ++j) {
      int m  = j + ((lane >> 4) << 3);
      int gr = bm0 + wm * 32 + t * 16 + m;
      if (gr < M) {
        float v = tanhf(acc[j] + bv) * wv;
        if (gr & 1) so += v; else se += v;
      }
    }
  }
  atomicAdd(&wsc[s * 2 + 0], se);
  atomicAdd(&wsc[s * 2 + 1], so);
}

// ---------------------------------------------------------------------------
// Elementwise / edge kernels
// ---------------------------------------------------------------------------
__global__ void cvt_bf16_k(const float* __restrict__ s,
                           unsigned short* __restrict__ d, int n) {
  int i = blockIdx.x * blockDim.x + threadIdx.x;
  if (i < n) d[i] = f2bf(s[i]);
}

__global__ void fill0_k(unsigned* __restrict__ p, int n) {
  int i = blockIdx.x * blockDim.x + threadIdx.x;
  if (i < n) p[i] = 0u;
}

// K in {5,3} encoders: out = lrelu(A @ W + b) into X cols [colOff, colOff+32)
__global__ void small_lin_k(const float* __restrict__ A, int K,
                            const float* __restrict__ W,
                            const float* __restrict__ b,
                            unsigned short* __restrict__ X, int colOff) {
  int i = blockIdx.x * blockDim.x + threadIdx.x;
  if (i >= NN * 32) return;
  int n = i >> 5, j = i & 31;
  float s = b[j];
  for (int k = 0; k < K; ++k) s += A[(size_t)n * K + k] * W[k * 32 + j];
  s = (s > 0.f) ? s : 0.01f * s;
  X[(size_t)n * DD + colOff + j] = f2bf(s);
}

// alpha[e,h] = (q[dst,h,:] . k[src,h,:]) / sqrt(D); atomicMax into mmax
__global__ void edge_alpha_k(const unsigned short* __restrict__ Q,
                             const unsigned short* __restrict__ Km,
                             const int* __restrict__ ei,
                             float* __restrict__ alpha,
                             unsigned* __restrict__ mmax) {
  int idx = blockIdx.x * blockDim.x + threadIdx.x;
  if (idx >= EE * 4) return;
  int e = idx >> 2, h = idx & 3;
  int src = ei[e], dst = ei[EE + e];
  const unsigned short* qp = Q + (size_t)dst * 512 + h * DD;
  const unsigned short* kp = Km + (size_t)src * 512 + h * DD;
  float s = 0.f;
  for (int d = 0; d < DD; d += 8) {
    uint4 qa = *(const uint4*)(qp + d);
    uint4 ka = *(const uint4*)(kp + d);
    const unsigned short* qu = (const unsigned short*)&qa;
    const unsigned short* ku = (const unsigned short*)&ka;
#pragma unroll
    for (int i = 0; i < 8; ++i) s += bf2f(qu[i]) * bf2f(ku[i]);
  }
  s *= 0.08838834764831845f;  // 1/sqrt(128)
  alpha[idx] = s;
  atomicMax(&mmax[dst * 4 + h], fenc(s));
}

// ex = exp(alpha - m[dst]); denom[dst] += ex
__global__ void edge_exp_k(const float* __restrict__ alpha,
                           const int* __restrict__ ei,
                           const unsigned* __restrict__ mmax,
                           float* __restrict__ exw,
                           float* __restrict__ den) {
  int idx = blockIdx.x * blockDim.x + threadIdx.x;
  if (idx >= EE * 4) return;
  int e = idx >> 2, h = idx & 3;
  int dst = ei[EE + e];
  unsigned enc = mmax[dst * 4 + h];
  float m = (enc == 0u) ? 0.f : fdec(enc);
  if (!(m == m) || fabsf(m) > 3.0e38f) m = 0.f;
  float ex = __expf(alpha[idx] - m);
  exw[idx] = ex;
  atomicAdd(&den[dst * 4 + h], ex);
}

// agg[dst,h,:] += (ex/denom) * v[src,h,:]; one block per edge, 128 lanes = D
__global__ __launch_bounds__(128) void edge_agg_k(
    const unsigned short* __restrict__ V,
    const float* __restrict__ exw, const float* __restrict__ den,
    const int* __restrict__ ei, float* __restrict__ agg) {
  int e = blockIdx.x;
  if (e >= EE) return;
  __shared__ float w4[4];
  int src = ei[e], dst = ei[EE + e];
  int t = threadIdx.x;
  if (t < 4) w4[t] = exw[e * 4 + t] / (den[dst * 4 + t] + 1e-16f);
  __syncthreads();
#pragma unroll
  for (int h = 0; h < 4; ++h) {
    float v = bf2f(V[(size_t)src * 512 + h * DD + t]);
    atomicAdd(&agg[((size_t)dst * 4 + h) * DD + t], w4[h] * v);
  }
}

// u = mean_h(agg) + skip
__global__ void node_u_k(const float* __restrict__ agg,
                         const float* __restrict__ skip,
                         unsigned short* __restrict__ U) {
  int i = blockIdx.x * blockDim.x + threadIdx.x;
  if (i >= NN * DD) return;
  int n = i >> 7, d = i & 127;
  const float* a = agg + (size_t)n * 512 + d;
  float u = 0.25f * (a[0] + a[128] + a[256] + a[384]) + skip[i];
  U[i] = f2bf(u);
}

__global__ void concat_ux_k(const unsigned short* __restrict__ U,
                            const unsigned short* __restrict__ X,
                            unsigned short* __restrict__ C) {
  int i = blockIdx.x * blockDim.x + threadIdx.x;
  if (i >= NN * DD) return;
  int n = i >> 7, d = i & 127;
  C[(size_t)n * 256 + d]       = U[i];
  C[(size_t)n * 256 + 128 + d] = X[i];
}

// z[:,mi,:] = tanh(u)*a + x*(1-a)
__global__ void emb_gate_k(const unsigned short* __restrict__ U,
                           const unsigned short* __restrict__ X,
                           const float* __restrict__ G,
                           unsigned short* __restrict__ Z, int mi) {
  int i = blockIdx.x * blockDim.x + threadIdx.x;
  if (i >= NN * DD) return;
  int n = i >> 7, d = i & 127;
  float a = G[i];
  float v = tanhf(bf2f(U[i])) * a + bf2f(X[i]) * (1.f - a);
  Z[((size_t)n * 2 + mi) * DD + d] = f2bf(v);
}

// beta = softmax_m(mean_n wsc); coeff[m] = mean_s beta[s,m]
__global__ void beta_coeff_k(const float* __restrict__ wsc,
                             float* __restrict__ coeff, float invN) {
  if (threadIdx.x == 0 && blockIdx.x == 0) {
    float c0 = 0.f, c1 = 0.f;
    for (int s = 0; s < 8; ++s) {
      float a  = wsc[s * 2 + 0] * invN;
      float b  = wsc[s * 2 + 1] * invN;
      float mx = fmaxf(a, b);
      float ea = __expf(a - mx), eb = __expf(b - mx);
      float d  = ea + eb;
      c0 += ea / d;
      c1 += eb / d;
    }
    coeff[0] = c0 * 0.125f;
    coeff[1] = c1 * 0.125f;
  }
}

// x_new = lrelu(coeff0*z0 + coeff1*z1)
__global__ void combine_k(const unsigned short* __restrict__ Z,
                          const float* __restrict__ coeff,
                          unsigned short* __restrict__ X) {
  int i = blockIdx.x * blockDim.x + threadIdx.x;
  if (i >= NN * DD) return;
  int n = i >> 7, d = i & 127;
  float v = coeff[0] * bf2f(Z[(size_t)(n * 2) * DD + d]) +
            coeff[1] * bf2f(Z[(size_t)(n * 2 + 1) * DD + d]);
  v = (v > 0.f) ? v : 0.01f * v;
  X[i] = f2bf(v);
}

// final 64 -> 2 projection
__global__ void out2_k(const float* __restrict__ O1,
                       const float* __restrict__ W,
                       const float* __restrict__ b,
                       float* __restrict__ out) {
  int n = blockIdx.x * blockDim.x + threadIdx.x;
  if (n >= NN) return;
  float s0 = b[0], s1 = b[1];
  for (int k = 0; k < 64; ++k) {
    float a = O1[(size_t)n * 64 + k];
    s0 += a * W[k * 2 + 0];
    s1 += a * W[k * 2 + 1];
  }
  out[n * 2 + 0] = s0;
  out[n * 2 + 1] = s1;
}

// ---------------------------------------------------------------------------
// Host orchestration
// ---------------------------------------------------------------------------
// Param order (JAX tree flatten, sorted dict keys):
enum {
  IN_PROP = 0, IN_CAT, IN_TWEET, IN_DES, IN_E0, IN_E1,
  P_INBOOL_B = 6, P_INBOOL_W, P_INDES_B, P_INDES_W, P_INNUM_B, P_INNUM_W,
  P_INTWEET_B, P_INTWEET_W, P_LIN1_B, P_LIN1_W, P_OUT1_B, P_OUT1_W,
  P_OUT2_B, P_OUT2_W,
  P_RGT1 = 20, P_RGT2 = 41
  // rgt block: +0 gate.b +1 gate.w +2 sem_b1 +3 sem_w1 +4 sem_w2
  //            +5..+12  tc0 {k.b,k.w,q.b,q.w,skip.b,skip.w,v.b,v.w}
  //            +13..+20 tc1 {same}
};

extern "C" void kernel_launch(void* const* d_in, const int* in_sizes, int n_in,
                              void* d_out, int out_size, void* d_ws,
                              size_t ws_size, hipStream_t stream) {
  (void)in_sizes; (void)n_in; (void)out_size; (void)ws_size;
  auto PF = [&](int i) { return (const float*)d_in[i]; };

  // ---- workspace carve (256B aligned) ----
  char* wp = (char*)d_ws;
  auto carve = [&](size_t bytes) -> char* {
    char* p = wp;
    wp += (bytes + 255) & ~(size_t)255;
    return p;
  };
  // bf16 weight pool layout (elements)
  const size_t OFF_WTWEET = 0, OFF_WDES = 24576, OFF_WLIN1 = 49152,
               OFF_WOUT1 = 65536, OFF_RGT0 = 73728;
  const size_t RGT_STRIDE = 589824, TC_STRIDE = 212992;
  const size_t WB_ELEMS = OFF_RGT0 + 2 * RGT_STRIDE;  // 1,253,376

  unsigned short* WBp  = (unsigned short*)carve(WB_ELEMS * 2);
  unsigned short* Xin  = (unsigned short*)carve((size_t)NN * DD * 2);
  unsigned short* Xbf  = (unsigned short*)carve((size_t)NN * DD * 2);
  unsigned short* Ubf  = (unsigned short*)carve((size_t)NN * DD * 2);
  unsigned short* CC   = (unsigned short*)carve((size_t)NN * 256 * 2);
  unsigned short* Qbf  = (unsigned short*)carve((size_t)NN * 512 * 2);
  unsigned short* Kbf  = (unsigned short*)carve((size_t)NN * 512 * 2);
  unsigned short* Vbf  = (unsigned short*)carve((size_t)NN * 512 * 2);
  float*          SKIP = (float*)carve((size_t)NN * DD * 4);
  float*          G    = (float*)carve((size_t)NN * DD * 4);
  unsigned short* Z    = (unsigned short*)carve((size_t)NN * 256 * 2);
  float*    ALPHA = (float*)carve((size_t)EE * 4 * 4);
  float*    EXW   = (float*)carve((size_t)EE * 4 * 4);
  unsigned* MMAX  = (unsigned*)carve((size_t)NN * 4 * 4);
  float*    DEN   = (float*)carve((size_t)NN * 4 * 4);
  float*    AGG   = (float*)carve((size_t)NN * 512 * 4);
  float*    WSC   = (float*)carve(16 * 4);
  float*    COEF  = (float*)carve(2 * 4);
  float*    O1    = (float*)carve((size_t)NN * 64 * 4);

  auto cvt = [&](const float* s, unsigned short* d, int n) {
    cvt_bf16_k<<<(n + 255) / 256, 256, 0, stream>>>(s, d, n);
  };
  auto zero = [&](void* p, int n) {
    fill0_k<<<(n + 255) / 256, 256, 0, stream>>>((unsigned*)p, n);
  };

  // ---- convert all weight matrices to bf16 (biases stay f32) ----
  cvt(PF(P_INTWEET_W), WBp + OFF_WTWEET, 24576);
  cvt(PF(P_INDES_W),   WBp + OFF_WDES,   24576);
  cvt(PF(P_LIN1_W),    WBp + OFF_WLIN1,  16384);
  cvt(PF(P_OUT1_W),    WBp + OFF_WOUT1,  8192);
  for (int r = 0; r < 2; ++r) {
    int pb = (r == 0) ? P_RGT1 : P_RGT2;
    unsigned short* wbR = WBp + OFF_RGT0 + r * RGT_STRIDE;
    cvt(PF(pb + 1), wbR + 0,     32768);   // gate.w (256x128)
    cvt(PF(pb + 3), wbR + 32768, 131072);  // sem_w1 (8x128x128)
    for (int i = 0; i < 2; ++i) {
      int tb = pb + 5 + i * 8;
      unsigned short* wtc = wbR + 163840 + i * TC_STRIDE;
      cvt(PF(tb + 3), wtc + 0,      65536);  // q.w
      cvt(PF(tb + 1), wtc + 65536,  65536);  // k.w
      cvt(PF(tb + 7), wtc + 131072, 65536);  // v.w
      cvt(PF(tb + 5), wtc + 196608, 16384);  // skip.w
    }
  }

  const int nd = NN * DD;
  const int eh = EE * 4;
  dim3 gNd((nd + 255) / 256), gEh((eh + 255) / 256);
  const int gM = (NN + BM - 1) / BM;       // 391
  const int gM2 = (2 * NN + BM - 1) / BM;  // 782

  // ---- input encoders -> Xin (bf16 N x 128) ----
  small_lin_k<<<(NN * 32 + 255) / 256, 256, 0, stream>>>(
      PF(IN_PROP), 5, PF(P_INNUM_W), PF(P_INNUM_B), Xin, 0);
  small_lin_k<<<(NN * 32 + 255) / 256, 256, 0, stream>>>(
      PF(IN_CAT), 3, PF(P_INBOOL_W), PF(P_INBOOL_B), Xin, 32);
  {
    dim3 g(gM, 1);
    gemm_wmma<float, 1, true><<<g, 256, 0, stream>>>(
        PF(IN_TWEET), 768, WBp + OFF_WTWEET, 32, PF(P_INTWEET_B),
        Xin, DD, 64, NN, 768, 32);
    gemm_wmma<float, 1, true><<<g, 256, 0, stream>>>(
        PF(IN_DES), 768, WBp + OFF_WDES, 32, PF(P_INDES_B),
        Xin, DD, 96, NN, 768, 32);
  }
  // ---- lin1: Xin -> Xbf ----
  {
    dim3 g(gM, DD / BN);
    gemm_wmma<unsigned short, 1, true><<<g, 256, 0, stream>>>(
        Xin, DD, WBp + OFF_WLIN1, DD, PF(P_LIN1_B), Xbf, DD, 0, NN, DD, DD);
  }

  // ---- two RGT layers ----
  for (int r = 0; r < 2; ++r) {
    int pb = (r == 0) ? P_RGT1 : P_RGT2;
    unsigned short* wbR = WBp + OFF_RGT0 + r * RGT_STRIDE;

    for (int i = 0; i < 2; ++i) {
      int tb = pb + 5 + i * 8;
      unsigned short* wtc = wbR + 163840 + i * TC_STRIDE;
      const int* ei = (const int*)d_in[IN_E0 + i];

      dim3 gQKV(gM, 512 / BN);
      gemm_wmma<unsigned short, 0, true><<<gQKV, 256, 0, stream>>>(
          Xbf, DD, wtc + 0, 512, PF(tb + 2), Qbf, 512, 0, NN, DD, 512);
      gemm_wmma<unsigned short, 0, true><<<gQKV, 256, 0, stream>>>(
          Xbf, DD, wtc + 65536, 512, PF(tb + 0), Kbf, 512, 0, NN, DD, 512);
      gemm_wmma<unsigned short, 0, true><<<gQKV, 256, 0, stream>>>(
          Xbf, DD, wtc + 131072, 512, PF(tb + 6), Vbf, 512, 0, NN, DD, 512);
      dim3 gSk(gM, DD / BN);
      gemm_wmma<unsigned short, 0, false><<<gSk, 256, 0, stream>>>(
          Xbf, DD, wtc + 196608, DD, PF(tb + 4), SKIP, DD, 0, NN, DD, DD);

      zero(MMAX, NN * 4);
      zero(DEN, NN * 4);
      zero(AGG, NN * 512);
      edge_alpha_k<<<gEh, 256, 0, stream>>>(Qbf, Kbf, ei, ALPHA, MMAX);
      edge_exp_k<<<gEh, 256, 0, stream>>>(ALPHA, ei, MMAX, EXW, DEN);
      edge_agg_k<<<EE, 128, 0, stream>>>(Vbf, EXW, DEN, ei, AGG);
      node_u_k<<<gNd, 256, 0, stream>>>(AGG, SKIP, Ubf);

      concat_ux_k<<<gNd, 256, 0, stream>>>(Ubf, Xbf, CC);
      dim3 gG(gM, DD / BN);
      gemm_wmma<unsigned short, 2, false><<<gG, 256, 0, stream>>>(
          CC, 256, wbR + 0, DD, PF(pb + 0), G, DD, 0, NN, 256, DD);
      emb_gate_k<<<gNd, 256, 0, stream>>>(Ubf, Xbf, G, Z, i);
    }

    // semantic attention (fused over 8 heads), then combine into Xbf
    zero(WSC, 16);
    {
      dim3 g(gM2, DD / BN, 8);
      gemm_sem<<<g, 256, 0, stream>>>(Z, wbR + 32768, PF(pb + 2), PF(pb + 4),
                                      WSC, 2 * NN);
    }
    beta_coeff_k<<<1, 32, 0, stream>>>(WSC, COEF, 1.0f / (float)NN);
    combine_k<<<gNd, 256, 0, stream>>>(Z, COEF, Xbf);
  }

  // ---- output head ----
  {
    dim3 g(gM, 64 / BN);
    gemm_wmma<unsigned short, 1, false><<<g, 256, 0, stream>>>(
        Xbf, DD, WBp + OFF_WOUT1, 64, PF(P_OUT1_B), O1, 64, 0, NN, DD, 64);
  }
  out2_k<<<(NN + 255) / 256, 256, 0, stream>>>(O1, PF(P_OUT2_W), PF(P_OUT2_B),
                                               (float*)d_out);
}
